// EmformerEncoder_62766652063948
// MI455X (gfx1250) — matrix operand order repaired and checked by reading, not compiled, not run on hardware
//
#include <hip/hip_runtime.h>
#include <hip/hip_bf16.h>

#define S_LEN  1272
#define S_PAD  1280
#define DMODEL 512
#define FFDIM  2048
#define NHEAD  8
#define DHEAD  64
#define NLAYER 8
#define NKB    (S_PAD / 32)   // 40 key blocks in attention
#define LN_EPS 1e-3f
#define NEG_BIG (-1e30f)

typedef __attribute__((ext_vector_type(16))) _Float16 v16h;
typedef __attribute__((ext_vector_type(8)))  _Float16 v8h;
typedef __attribute__((ext_vector_type(8)))  float    v8f;

union AFrag { v16h v; v8h h[2]; };

__device__ __forceinline__ v8f wmma_f16(v16h a, v16h b, v8f c) {
  // D = A(16x32 f16) * B(32x16 f16) + C(16x16 f32)
  return __builtin_amdgcn_wmma_f32_16x16x32_f16(false, a, false, b, (short)0, c,
                                                false, false);
}

// Analytic Emformer full-context mask (T_MAX=1024, CHUNK=32, LEFT=20, RIGHT=8).
// Rows/cols 0..247: hard-copied right contexts (8 per chunk, chunks 0..30);
// rows/cols 248..1271: body frames.
__device__ __forceinline__ bool emf_mask(int a, int b) {
  if (a < 248) {
    int i = a >> 3;
    if (b < 248) return (b >> 3) == i;
    int t = b - 248;
    int lo = 32 * i - 20; lo = lo < 0 ? 0 : lo;
    return (t >= lo) && (t < 32 * (i + 1));
  } else {
    int t = a - 248;
    int c = t >> 5;
    if (b < 248) return (c <= 30) && ((b >> 3) == c);
    int u = b - 248;
    int lo = 32 * c - 20; lo = lo < 0 ? 0 : lo;
    return (u >= lo) && (u < 32 * (c + 1));
  }
}

// Convert f32 weight W[K][N] into f16 WMMA B-fragment layout:
// frag(kt,nt) is 512 halves; element (lane = n+16g, e) = W[kt*32+16g+e][nt*16+n].
// One thread produces one lane's 16 contiguous halves (32B store).
__global__ __launch_bounds__(256) void swizzle_w_kernel(
    const float* __restrict__ W, _Float16* __restrict__ Wf, int K, int N) {
  const int idx = blockIdx.x * 256 + threadIdx.x;   // [0, K*N/16)
  const int lane = idx & 31;
  const int tile = idx >> 5;
  const int ntiles = N >> 4;
  const int nt = tile % ntiles;
  const int kt = tile / ntiles;
  const int g = lane >> 4, n = lane & 15;
  const float* src = W + (size_t)(kt * 32 + 16 * g) * N + nt * 16 + n;
  v8h d0, d1;
  #pragma unroll
  for (int e = 0; e < 8; ++e) d0[e] = (_Float16)src[(size_t)e * N];
  #pragma unroll
  for (int e = 0; e < 8; ++e) d1[e] = (_Float16)src[(size_t)(e + 8) * N];
  v8h* dst = (v8h*)(Wf + (size_t)idx * 16);
  dst[0] = d0;
  dst[1] = d1;
}

// s = a (+ b); optional sum_out = s; LN(s)*gamma+beta -> out32 and/or out16.
__global__ __launch_bounds__(256) void add_ln_kernel(
    const float* __restrict__ a, const float* __restrict__ b,
    const float* __restrict__ gamma, const float* __restrict__ beta,
    float* __restrict__ sum_out, float* __restrict__ out32,
    _Float16* __restrict__ out16) {
  __shared__ float wred[8];
  const int row = blockIdx.x;
  const int tid = threadIdx.x;
  const int lane = tid & 31, wv = tid >> 5;
  const size_t base = (size_t)row * DMODEL;
  float s0 = a[base + tid], s1 = a[base + tid + 256];
  if (b) { s0 += b[base + tid]; s1 += b[base + tid + 256]; }
  if (sum_out) { sum_out[base + tid] = s0; sum_out[base + tid + 256] = s1; }
  float p = s0 + s1;
  p += __shfl_xor(p, 1, 32);  p += __shfl_xor(p, 2, 32);
  p += __shfl_xor(p, 4, 32);  p += __shfl_xor(p, 8, 32);
  p += __shfl_xor(p, 16, 32);
  if (lane == 0) wred[wv] = p;
  __syncthreads();
  float tot = 0.0f;
  #pragma unroll
  for (int i = 0; i < 8; ++i) tot += wred[i];
  const float mu = tot * (1.0f / DMODEL);
  __syncthreads();
  const float d0 = s0 - mu, d1 = s1 - mu;
  p = d0 * d0 + d1 * d1;
  p += __shfl_xor(p, 1, 32);  p += __shfl_xor(p, 2, 32);
  p += __shfl_xor(p, 4, 32);  p += __shfl_xor(p, 8, 32);
  p += __shfl_xor(p, 16, 32);
  if (lane == 0) wred[wv] = p;
  __syncthreads();
  tot = 0.0f;
  #pragma unroll
  for (int i = 0; i < 8; ++i) tot += wred[i];
  const float inv = rsqrtf(tot * (1.0f / DMODEL) + LN_EPS);
  const float y0 = d0 * inv * gamma[tid] + beta[tid];
  const float y1 = d1 * inv * gamma[tid + 256] + beta[tid + 256];
  if (out32) { out32[base + tid] = y0; out32[base + tid + 256] = y1; }
  if (out16) { out16[base + tid] = (_Float16)y0; out16[base + tid + 256] = (_Float16)y1; }
}

// Core: one wave computes a 32x64 output tile (two 16-row tiles sharing the
// four B-fragments) accumulating over K. acc[0..3]: rows m0..m0+15;
// acc[4..7]: rows m0+16..m0+31; tile t covers cols nb+16t..nb+16t+15.
__device__ __forceinline__ void gemm_core(
    const _Float16* __restrict__ A, const _Float16* __restrict__ Wf,
    int K, int ntiles, int m0, int nb, int lane, v8f acc[8]) {
  const int g = lane >> 4, hl = lane & 15;
  const _Float16* arow0 = A + (size_t)(m0 + hl) * K;
  const _Float16* arow1 = arow0 + (size_t)16 * K;
  const _Float16* wbase = Wf + ((size_t)(nb >> 4) * 32 + lane) * 16;
  #pragma unroll 2
  for (int k0 = 0; k0 < K; k0 += 32) {
    AFrag a0, a1;
    a0.h[0] = *(const v8h*)(arow0 + k0 + 16 * g);      // K = 16g..16g+7
    a0.h[1] = *(const v8h*)(arow0 + k0 + 16 * g + 8);  // K = 16g+8..16g+15
    a1.h[0] = *(const v8h*)(arow1 + k0 + 16 * g);
    a1.h[1] = *(const v8h*)(arow1 + k0 + 16 * g + 8);
    const _Float16* wp = wbase + (size_t)(k0 >> 5) * ntiles * 512;
    #pragma unroll
    for (int t = 0; t < 4; ++t) {
      v16h b = *(const v16h*)(wp + t * 512);
      acc[t]     = wmma_f16(a0.v, b, acc[t]);
      acc[4 + t] = wmma_f16(a1.v, b, acc[4 + t]);
    }
  }
}

// Epilogue: (acc + bias) * scale (+ resid); store f16 row-major, f32, or
// (vmode==1) f16 in the attention-V swizzled B-fragment layout.
__device__ __forceinline__ void gemm_store(
    const v8f acc[8], const float* __restrict__ bias,
    const float* __restrict__ resid, float scale,
    _Float16* __restrict__ out16, float* __restrict__ out32,
    int N, int m0, int nb, int lane, int vmode) {
  const int g = lane >> 4, hl = lane & 15;
  #pragma unroll
  for (int half = 0; half < 2; ++half) {
    const int mb = m0 + 16 * half;
    #pragma unroll
    for (int t = 0; t < 4; ++t) {
      const int col = nb + t * 16 + hl;
      const float bcol = bias[col];
      #pragma unroll
      for (int r = 0; r < 8; ++r) {                    // C: M = r+8g, N = hl
        const int row = mb + r + 8 * g;
        if (row < S_LEN) {
          float val = (acc[half * 4 + t][r] + bcol) * scale;
          if (resid) val += resid[(size_t)row * N + col];
          if (vmode == 1) {
            // V swizzle: [head][key>>5][dk>>4][(dk&15)+16*((key>>4)&1)][key&15]
            const int head = col >> 6, dkl = col & 63;
            const size_t o =
                ((size_t)((head * NKB + (row >> 5)) * 4 + (dkl >> 4)) * 512)
                + (size_t)((dkl & 15) + 16 * ((row >> 4) & 1)) * 16
                + (size_t)(row & 15);
            out16[o] = (_Float16)val;
          } else if (out16) {
            out16[(size_t)row * N + col] = (_Float16)val;
          } else {
            out32[(size_t)row * N + col] = val;
          }
        }
      }
    }
  }
}

// Generic GEMM: block = 256 threads = 8 waves; wave owns 32 rows x 64 cols.
__global__ __launch_bounds__(256) void gemm_wmma_kernel(
    const _Float16* __restrict__ A, const _Float16* __restrict__ Wf,
    const float* __restrict__ bias, const float* __restrict__ resid,
    float scale, _Float16* __restrict__ out16, float* __restrict__ out32,
    int K, int N, int vmode) {
  const int wv = threadIdx.x >> 5, lane = threadIdx.x & 31;
  const int m0 = (blockIdx.x * 8 + wv) * 32;
  const int nb = blockIdx.y * 64;
  v8f acc[8] = {};
  gemm_core(A, Wf, K, N >> 4, m0, nb, lane, acc);
  gemm_store(acc, bias, resid, scale, out16, out32, N, m0, nb, lane, vmode);
}

// Fused Q/K/V projection: blockIdx.z selects the projection so all three run
// in a single launch (3x machine fill vs serialized launches on one stream).
__global__ __launch_bounds__(256) void qkv_wmma_kernel(
    const _Float16* __restrict__ A,
    const _Float16* __restrict__ Wqf, const _Float16* __restrict__ Wkf,
    const _Float16* __restrict__ Wvf,
    const float* __restrict__ bq, const float* __restrict__ bk,
    const float* __restrict__ bv,
    _Float16* __restrict__ q16, _Float16* __restrict__ k16,
    _Float16* __restrict__ vswz) {
  const int wv = threadIdx.x >> 5, lane = threadIdx.x & 31;
  const int m0 = (blockIdx.x * 8 + wv) * 32;
  const int nb = blockIdx.y * 64;
  const int z = blockIdx.z;
  const _Float16* Wf = (z == 0) ? Wqf : (z == 1) ? Wkf : Wvf;
  const float* bias  = (z == 0) ? bq  : (z == 1) ? bk  : bv;
  _Float16* out      = (z == 0) ? q16 : (z == 1) ? k16 : vswz;
  const float scale  = (z == 0) ? 0.125f : 1.0f;   // 1/sqrt(dk) folded into q
  const int vmode    = (z == 2) ? 1 : 0;
  v8f acc[8] = {};
  gemm_core(A, Wf, DMODEL, DMODEL >> 4, m0, nb, lane, acc);
  gemm_store(acc, bias, nullptr, scale, out, nullptr, DMODEL, m0, nb, lane, vmode);
}

// Flash attention: block = 128 threads = 4 waves; wave owns one 16-query tile
// of one head; streams 32-key blocks with online softmax. q pre-scaled by
// 1/sqrt(dk). K read row-major (already B^T layout); V read from swizzled
// fragment layout; only P crosses LDS (C-layout -> A-layout transpose).
__global__ __launch_bounds__(128) void attn_kernel(
    const _Float16* __restrict__ q, const _Float16* __restrict__ kx,
    const _Float16* __restrict__ vswz, float* __restrict__ out) {
  __shared__ __align__(32) _Float16 Ps[4][16][32];
  const int tid = threadIdx.x;
  const int wv = tid >> 5, lane = tid & 31;
  const int g = lane >> 4, hl = lane & 15;
  const int q0 = (blockIdx.x * 4 + wv) * 16;
  const int head = blockIdx.y;
  const int hoff = head * DHEAD;

  AFrag aq0, aq1;                                      // Q[16 x 64]
  {
    const _Float16* qr = q + (size_t)(q0 + hl) * DMODEL + hoff;
    aq0.h[0] = *(const v8h*)(qr + 16 * g);
    aq0.h[1] = *(const v8h*)(qr + 8 + 16 * g);
    aq1.h[0] = *(const v8h*)(qr + 32 + 16 * g);
    aq1.h[1] = *(const v8h*)(qr + 40 + 16 * g);
  }
  v8f acc0 = {}, acc1 = {}, acc2 = {}, acc3 = {};
  float mrow[8], lrow[8];
  #pragma unroll
  for (int r = 0; r < 8; ++r) { mrow[r] = NEG_BIG; lrow[r] = 0.0f; }

  for (int kb = 0; kb < NKB; ++kb) {
    const int kbase = kb * 32;
    v8f s[2];
    #pragma unroll
    for (int kt = 0; kt < 2; ++kt) {                   // scores: Q @ K^T
      const int k0 = kbase + kt * 16;
      const _Float16* kr = kx + (size_t)(k0 + hl) * DMODEL + hoff;
      v16h b0 = *(const v16h*)(kr + 16 * g);           // dk 0..31
      v16h b1 = *(const v16h*)(kr + 32 + 16 * g);      // dk 32..63
      v8f c = {};
      c = wmma_f16(aq0.v, b0, c);
      c = wmma_f16(aq1.v, b1, c);
      const int kj = k0 + hl;
      #pragma unroll
      for (int r = 0; r < 8; ++r) {
        const int qi = q0 + r + 8 * g;
        const bool ok = (qi < S_LEN) && (kj < S_LEN) && emf_mask(qi, kj);
        if (!ok) c[r] = NEG_BIG;
      }
      s[kt] = c;
    }
    #pragma unroll
    for (int r = 0; r < 8; ++r) {                      // online softmax
      float x0 = s[0][r], x1 = s[1][r];
      float t = fmaxf(x0, x1);
      t = fmaxf(t, __shfl_xor(t, 1, 32));
      t = fmaxf(t, __shfl_xor(t, 2, 32));
      t = fmaxf(t, __shfl_xor(t, 4, 32));
      t = fmaxf(t, __shfl_xor(t, 8, 32));
      const float mnew = fmaxf(mrow[r], t);
      const float p0 = __expf(x0 - mnew), p1 = __expf(x1 - mnew);
      float rs = p0 + p1;
      rs += __shfl_xor(rs, 1, 32);
      rs += __shfl_xor(rs, 2, 32);
      rs += __shfl_xor(rs, 4, 32);
      rs += __shfl_xor(rs, 8, 32);
      const float corr = __expf(mrow[r] - mnew);
      lrow[r] = lrow[r] * corr + rs;
      mrow[r] = mnew;
      acc0[r] *= corr; acc1[r] *= corr; acc2[r] *= corr; acc3[r] *= corr;
      Ps[wv][r + 8 * g][hl]      = (_Float16)p0;       // C-layout -> LDS
      Ps[wv][r + 8 * g][16 + hl] = (_Float16)p1;
    }
    __syncthreads();
    AFrag ap;                                          // reload P in A-layout
    ap.h[0] = *(const v8h*)&Ps[wv][hl][16 * g];
    ap.h[1] = *(const v8h*)&Ps[wv][hl][16 * g + 8];
    const _Float16* vp = vswz + (size_t)((head * NKB + kb) * 4) * 512 + lane * 16;
    v16h bv0 = *(const v16h*)(vp);
    v16h bv1 = *(const v16h*)(vp + 512);
    v16h bv2 = *(const v16h*)(vp + 1024);
    v16h bv3 = *(const v16h*)(vp + 1536);
    acc0 = wmma_f16(ap.v, bv0, acc0);
    acc1 = wmma_f16(ap.v, bv1, acc1);
    acc2 = wmma_f16(ap.v, bv2, acc2);
    acc3 = wmma_f16(ap.v, bv3, acc3);
    __syncthreads();
  }
  #pragma unroll
  for (int r = 0; r < 8; ++r) {
    const int qi = q0 + r + 8 * g;
    if (qi < S_LEN) {
      const float invl = 1.0f / lrow[r];
      const size_t o = (size_t)qi * DMODEL + hoff + hl;
      out[o]      = acc0[r] * invl;
      out[o + 16] = acc1[r] * invl;
      out[o + 32] = acc2[r] * invl;
      out[o + 48] = acc3[r] * invl;
    }
  }
}

extern "C" void kernel_launch(void* const* d_in, const int* in_sizes, int n_in,
                              void* d_out, int out_size, void* d_ws, size_t ws_size,
                              hipStream_t stream) {
  (void)in_sizes; (void)n_in; (void)out_size; (void)ws_size;
  const float* x   = (const float*)d_in[0];
  // d_in[1] (mask) recomputed analytically on device.
  const float* gin = (const float*)d_in[2];
  const float* bin = (const float*)d_in[3];
  const float* wq  = (const float*)d_in[4];
  const float* bq  = (const float*)d_in[5];
  const float* wk  = (const float*)d_in[6];
  const float* bk  = (const float*)d_in[7];
  const float* wv  = (const float*)d_in[8];
  const float* bv  = (const float*)d_in[9];
  const float* g1  = (const float*)d_in[10];
  const float* b1  = (const float*)d_in[11];
  const float* w1  = (const float*)d_in[12];
  const float* bb1 = (const float*)d_in[13];
  const float* w2  = (const float*)d_in[14];
  const float* bb2 = (const float*)d_in[15];
  const float* g2  = (const float*)d_in[16];
  const float* b2  = (const float*)d_in[17];

  char* ws = (char*)d_ws;
  size_t off = 0;
  auto take = [&](size_t bytes) {
    char* p = ws + off;
    off += (bytes + 255) & ~(size_t)255;
    return p;
  };
  float*    hn32   = (float*)take((size_t)S_PAD * DMODEL * 4);
  float*    attn32 = (float*)take((size_t)S_PAD * DMODEL * 4);
  float*    res32  = (float*)take((size_t)S_PAD * DMODEL * 4);
  float*    ff32   = (float*)take((size_t)S_PAD * DMODEL * 4);
  float*    hbuf   = (float*)take((size_t)S_PAD * DMODEL * 4);
  _Float16* hn16   = (_Float16*)take((size_t)S_PAD * DMODEL * 2);
  _Float16* q16    = (_Float16*)take((size_t)S_PAD * DMODEL * 2);
  _Float16* k16    = (_Float16*)take((size_t)S_PAD * DMODEL * 2);
  _Float16* vswz   = (_Float16*)take((size_t)S_PAD * DMODEL * 2);
  _Float16* y16    = (_Float16*)take((size_t)S_PAD * DMODEL * 2);
  _Float16* z16    = (_Float16*)take((size_t)S_PAD * FFDIM * 2);
  _Float16* wqf    = (_Float16*)take((size_t)DMODEL * DMODEL * 2);
  _Float16* wkf    = (_Float16*)take((size_t)DMODEL * DMODEL * 2);
  _Float16* wvf    = (_Float16*)take((size_t)DMODEL * DMODEL * 2);
  _Float16* w1f    = (_Float16*)take((size_t)DMODEL * FFDIM * 2);
  _Float16* w2f    = (_Float16*)take((size_t)FFDIM * DMODEL * 2);

  const int BLK_DD = DMODEL * DMODEL / 16 / 256;   // 64
  const int BLK_DF = DMODEL * FFDIM / 16 / 256;    // 256

  for (int l = 0; l < NLAYER; ++l) {
    // Weight -> f16 fragment-layout conversion for this layer.
    swizzle_w_kernel<<<BLK_DD, 256, 0, stream>>>(wq + (size_t)l * DMODEL * DMODEL, wqf, DMODEL, DMODEL);
    swizzle_w_kernel<<<BLK_DD, 256, 0, stream>>>(wk + (size_t)l * DMODEL * DMODEL, wkf, DMODEL, DMODEL);
    swizzle_w_kernel<<<BLK_DD, 256, 0, stream>>>(wv + (size_t)l * DMODEL * DMODEL, wvf, DMODEL, DMODEL);
    swizzle_w_kernel<<<BLK_DF, 256, 0, stream>>>(w1 + (size_t)l * DMODEL * FFDIM, w1f, DMODEL, FFDIM);
    swizzle_w_kernel<<<BLK_DF, 256, 0, stream>>>(w2 + (size_t)l * FFDIM * DMODEL, w2f, FFDIM, DMODEL);

    const float* hin = (l == 0) ? x : hbuf;
    // hn = LN_in(h): f32 for residual + f16 matmul operand
    add_ln_kernel<<<S_LEN, 256, 0, stream>>>(hin, nullptr, gin + l * DMODEL,
                                             bin + l * DMODEL, nullptr, hn32, hn16);
    // fused q,k,v projections (z dim); v stored swizzled for PV fragments
    qkv_wmma_kernel<<<dim3(5, 8, 3), 256, 0, stream>>>(
        hn16, wqf, wkf, wvf, bq + l * DMODEL, bk + l * DMODEL, bv + l * DMODEL,
        q16, k16, vswz);
    // masked flash attention
    attn_kernel<<<dim3(20, NHEAD), 128, 0, stream>>>(q16, k16, vswz, attn32);
    // attn_out = attn + hn; y = LN1(attn_out)
    add_ln_kernel<<<S_LEN, 256, 0, stream>>>(attn32, hn32, g1 + l * DMODEL,
                                             b1 + l * DMODEL, res32, nullptr, y16);
    // FFN (no nonlinearity): z = y@W1+b1 ; f = z@W2+b2 + attn_out
    gemm_wmma_kernel<<<dim3(5, 32), 256, 0, stream>>>(
        y16, w1f, bb1 + l * FFDIM, nullptr, 1.0f, z16, nullptr, DMODEL, FFDIM, 0);
    gemm_wmma_kernel<<<dim3(5, 8), 256, 0, stream>>>(
        z16, w2f, bb2 + l * DMODEL, res32, 1.0f, nullptr, ff32, FFDIM, DMODEL, 0);
    // h_next = LN2(f)
    float* hdst = (l == NLAYER - 1) ? (float*)d_out : hbuf;
    add_ln_kernel<<<S_LEN, 256, 0, stream>>>(ff32, nullptr, g2 + l * DMODEL,
                                             b2 + l * DMODEL, nullptr, hdst, nullptr);
  }
}
